// Non_local_block_multi_head_31619549233972
// MI455X (gfx1250) — matrix operand order repaired and checked by reading, not compiled, not run on hardware
//
#include <hip/hip_runtime.h>

// ---------------------------------------------------------------------------
// CDNA5 (gfx1250) wave32 WMMA implementation of the multi-head non-local block.
// Heavy GEMMs use v_wmma_f32_16x16x32_f16 (f16 in, f32 accumulate).
// Flash-attention kernel stages shared Q/V tiles via global_load_async_to_lds.
// ---------------------------------------------------------------------------

typedef __attribute__((ext_vector_type(16))) _Float16 v16h;
typedef __attribute__((ext_vector_type(8)))  float    v8f;

#define WMMA(A, B, C) __builtin_amdgcn_wmma_f32_16x16x32_f16( \
    false, (A), false, (B), (short)0, (C), false, false)

// Problem constants
#define BATCH 8
#define NHEAD 8
#define HH 32
#define WW 32
#define NPOS 1024          // HH*WW
#define CIN 512
#define DHEAD 64
#define HOUT 64
#define COUT 512
#define BN_EPS 1e-3f

// ---- Fragment loaders (layouts per CDNA5 ISA 7.12.2, wave32) ---------------

// A-matrix 16x32 f16: lane = kg*16 + m (m=row). elems 0..7 -> K=8*kg+e,
// elems 8..15 -> K=16+8*kg+(e-8). Source: row-major [16, ld] at base.
__device__ __forceinline__ v16h frag_a(const _Float16* base, int ld) {
    const int lane = threadIdx.x & 31;
    const int m = lane & 15, kg = lane >> 4;
    const _Float16* p = base + m * ld + 8 * kg;
    v16h r;
    *reinterpret_cast<uint4*>(&r)       = *reinterpret_cast<const uint4*>(p);
    *(reinterpret_cast<uint4*>(&r) + 1) = *reinterpret_cast<const uint4*>(p + 16);
    return r;
}

// B-matrix 32x16 f16: lane = kh*16 + n (n=col). elems 0..7 -> K=8*kh+e, elems 8..15 -> K=16+8*kh+(e-8).
// Source: K-contiguous "transposed" layout (col n at base + n*ld).
__device__ __forceinline__ v16h frag_bt(const _Float16* base, int ld) {
    const int lane = threadIdx.x & 31;
    const int nc = lane & 15, kh = lane >> 4;
    const _Float16* p = base + nc * ld + 8 * kh;
    v16h r;
    *reinterpret_cast<uint4*>(&r)       = *reinterpret_cast<const uint4*>(p);
    *(reinterpret_cast<uint4*>(&r) + 1) = *reinterpret_cast<const uint4*>(p + 16);
    return r;
}

// CDNA5 async copy: 16 bytes per lane, global -> LDS, tracked by ASYNCcnt.
__device__ __forceinline__ void async_copy_b128(unsigned lds_addr, const void* gptr) {
    asm volatile("global_load_async_to_lds_b128 %0, %1, off"
                 :: "v"(lds_addr), "v"((unsigned long long)(size_t)gptr)
                 : "memory");
}

// Reductions across the 16 lanes that share one C/D-matrix row group.
__device__ __forceinline__ float rowmax16(float v) {
#pragma unroll
    for (int off = 1; off < 16; off <<= 1) v = fmaxf(v, __shfl_xor(v, off, 32));
    return v;
}
__device__ __forceinline__ float rowsum16(float v) {
#pragma unroll
    for (int off = 1; off < 16; off <<= 1) v += __shfl_xor(v, off, 32);
    return v;
}

// ---- Prep kernels ----------------------------------------------------------

__global__ void cvt_f32_f16(const float* __restrict__ in, _Float16* __restrict__ out, int n) {
    int i = blockIdx.x * blockDim.x + threadIdx.x;
    if (i < n) out[i] = (_Float16)in[i];
}

// out[g][d][c] = (f16) in[g][c][d]   (makes every WMMA B-load K-contiguous)
__global__ void transpose_cd(const float* __restrict__ in, _Float16* __restrict__ out,
                             int G, int C, int D) {
    int i = blockIdx.x * blockDim.x + threadIdx.x;
    int n = G * C * D;
    if (i >= n) return;
    int c = i % C;
    int d = (i / C) % D;
    int g = i / (C * D);
    out[i] = (_Float16)in[(size_t)g * C * D + (size_t)c * D + d];
}

__global__ void bn_prep(const float* __restrict__ gamma, const float* __restrict__ beta,
                        const float* __restrict__ mean, const float* __restrict__ var,
                        float* __restrict__ scale, float* __restrict__ bias) {
    int i = blockIdx.x * blockDim.x + threadIdx.x;
    if (i < NHEAD * HOUT) {
        float s = gamma[i] * rsqrtf(var[i] + BN_EPS);
        scale[i] = s;
        bias[i] = beta[i] - mean[i] * s;
    }
}

// ---- Q/K/V projection: X[b](1024x512) @ W[h](512x64), WMMA ------------------
// Q,K stored row-major [bh][n][64]; V stored transposed [bh][64][n].
__global__ __launch_bounds__(256) void qkv_proj(
    const _Float16* __restrict__ xh,
    const _Float16* __restrict__ WqT, const _Float16* __restrict__ WkT,
    const _Float16* __restrict__ WvT,
    _Float16* __restrict__ Qh, _Float16* __restrict__ Kh, _Float16* __restrict__ Vt) {
    const int wv = threadIdx.x >> 5;
    const int lane = threadIdx.x & 31;
    const int nc = lane & 15, hi = lane >> 4;
    int bid = blockIdx.x;
    const int rowtile = bid & 7;  bid >>= 3;
    const int which = bid % 3;    bid /= 3;
    const int h = bid & 7, b = bid >> 3;
    const int rowbase = rowtile * 128 + wv * 16;

    const _Float16* X = xh + (size_t)b * NPOS * CIN;
    const _Float16* WT = (which == 0 ? WqT : which == 1 ? WkT : WvT) + (size_t)h * DHEAD * CIN;

    v8f acc[4] = {v8f{}, v8f{}, v8f{}, v8f{}};
    for (int k = 0; k < CIN; k += 32) {
        v16h a = frag_a(X + (size_t)rowbase * CIN + k, CIN);
#pragma unroll
        for (int t = 0; t < 4; t++) {
            v16h bf = frag_bt(WT + (size_t)(t * 16) * CIN + k, CIN);
            acc[t] = WMMA(a, bf, acc[t]);
        }
    }
    const int bh = b * NHEAD + h;
    if (which < 2) {
        _Float16* out = (which == 0 ? Qh : Kh) + (size_t)bh * NPOS * DHEAD;
#pragma unroll
        for (int t = 0; t < 4; t++)
#pragma unroll
            for (int r = 0; r < 8; r++)
                out[(size_t)(rowbase + r + 8 * hi) * DHEAD + t * 16 + nc] = (_Float16)acc[t][r];
    } else {
        _Float16* out = Vt + (size_t)bh * DHEAD * NPOS;
#pragma unroll
        for (int t = 0; t < 4; t++)
#pragma unroll
            for (int r = 0; r < 8; r++)
                out[(size_t)(t * 16 + nc) * NPOS + rowbase + r + 8 * hi] = (_Float16)acc[t][r];
    }
}

// ---- Relative position logits (rel_to_abs collapsed to direct indexing) ----
__global__ void rel_logits(const _Float16* __restrict__ Qh, const float* __restrict__ rel,
                           float* __restrict__ out, int use_x) {
    int i = blockIdx.x * blockDim.x + threadIdx.x;
    if (i >= BATCH * NHEAD * NPOS * 32) return;
    int t = i & 31;
    int pos = (i >> 5) & (NPOS - 1);
    int bh = i >> 15;
    int h = bh & (NHEAD - 1);
    int coord = use_x ? (pos >> 5) : (pos & 31);
    int m = t - coord + 31;  // in [0, 62]
    const _Float16* q = Qh + ((size_t)bh * NPOS + pos) * DHEAD;
    const float* rw = rel + ((size_t)h * 63 + m) * DHEAD;
    float s = 0.f;
#pragma unroll 8
    for (int d = 0; d < DHEAD; d++) s += (float)q[d] * rw[d];
    out[i] = s;
}

// ---- Flash attention -------------------------------------------------------
// Per wave: 16 key-rows, online softmax over 32-column chunks. Q/V chunk data
// is identical for all 8 waves -> staged once per block into LDS with
// global_load_async_to_lds_b128 (ASYNCcnt), then consumed as WMMA B-fragments.
__global__ __launch_bounds__(256) void attn_flash(
    const _Float16* __restrict__ Qh, const _Float16* __restrict__ Kh,
    const _Float16* __restrict__ Vt, const float* __restrict__ wabs,
    const float* __restrict__ habs, _Float16* __restrict__ Oh) {
    __shared__ __align__(16) _Float16 qbuf[32][64];     // Q rows jb..jb+31
    __shared__ __align__(16) _Float16 vbuf[64][32];     // Vt rows d=0..63, cols jb..jb+31
    __shared__ __align__(16) _Float16 pbuf[8][16][32];  // per-wave P tile
    const int wv = threadIdx.x >> 5;
    const int lane = threadIdx.x & 31;
    const int nc = lane & 15, hi = lane >> 4;
    const int bh = blockIdx.x >> 3;
    const int rowbase = (blockIdx.x & 7) * 128 + wv * 16;

    const _Float16* Qhh = Qh + (size_t)bh * NPOS * DHEAD;
    const _Float16* Khh = Kh + (size_t)bh * NPOS * DHEAD;
    const _Float16* Vth = Vt + (size_t)bh * DHEAD * NPOS;
    const float* wab = wabs + (size_t)bh * NPOS * 32;
    const float* hab = habs + (size_t)bh * NPOS * 32;

    // K-row A-fragments (d = 0..31, 32..63), reused over all column chunks.
    v16h ka0 = frag_a(Khh + (size_t)rowbase * DHEAD + 0, DHEAD);
    v16h ka1 = frag_a(Khh + (size_t)rowbase * DHEAD + 32, DHEAD);

    // Cooperative staging slices (constant per thread across chunks)
    const int qrow = 4 * wv + (lane >> 3), qseg = lane & 7;   // 32 rows x 8 segs
    const int vrow = 8 * wv + (lane >> 2), vseg = lane & 3;   // 64 rows x 4 segs (4 segs of 8 halfs = 32)
    const unsigned qlds = (unsigned)(size_t)&qbuf[qrow][qseg * 8];
    const unsigned vlds = (unsigned)(size_t)&vbuf[vrow][vseg * 8];

    int irow[8];
    float ww0[8], ww1[8];
#pragma unroll
    for (int r = 0; r < 8; r++) {
        irow[r] = rowbase + r + 8 * hi;
        ww0[r] = wab[irow[r] * 32 + nc];        // w-bias, cols jlo = nc
        ww1[r] = wab[irow[r] * 32 + 16 + nc];   // w-bias, cols jlo = 16+nc
    }

    v8f o0 = {}, o1 = {}, o2 = {}, o3 = {};
    float mrun[8], lrun[8];
#pragma unroll
    for (int r = 0; r < 8; r++) { mrun[r] = -3.0e38f; lrun[r] = 0.f; }

    for (int jb = 0; jb < NPOS; jb += 32) {
        const int jc = jb >> 5;
        // ---- async stage Q[jb..jb+32) and Vt[:, jb..jb+32) into LDS --------
        async_copy_b128(qlds, Qhh + (size_t)(jb + qrow) * DHEAD + qseg * 8);
        async_copy_b128(vlds, Vth + (size_t)vrow * NPOS + jb + vseg * 8);
        asm volatile("s_wait_asynccnt 0" ::: "memory");
        __syncthreads();

        if (jb + 32 < NPOS) {  // uniform; warm L2 for the next chunk
            __builtin_prefetch(Qhh + (size_t)(jb + 32) * DHEAD, 0, 1);
            __builtin_prefetch(Vth + jb + 32, 0, 1);
        }

        // ---- scores S[16 x 32] = K_rows . Q_cols^T -------------------------
        v8f s0 = {}, s1 = {};
        s0 = WMMA(ka0, frag_bt(&qbuf[0][0], 64), s0);
        s0 = WMMA(ka1, frag_bt(&qbuf[0][32], 64), s0);
        s1 = WMMA(ka0, frag_bt(&qbuf[16][0], 64), s1);
        s1 = WMMA(ka1, frag_bt(&qbuf[16][32], 64), s1);

        float p0[8], p1[8];
#pragma unroll
        for (int r = 0; r < 8; r++) {
            float hb = hab[irow[r] * 32 + jc];
            float a = s0[r] + hb + ww0[r];
            float bq = s1[r] + hb + ww1[r];
            float cm = rowmax16(fmaxf(a, bq));
            float mn = fmaxf(mrun[r], cm);
            float sc = __expf(mrun[r] - mn);
            float e0 = __expf(a - mn);
            float e1 = __expf(bq - mn);
            lrun[r] = lrun[r] * sc + rowsum16(e0 + e1);
            mrun[r] = mn;
            o0[r] *= sc; o1[r] *= sc; o2[r] *= sc; o3[r] *= sc;
            p0[r] = e0; p1[r] = e1;
        }
        // D-layout -> A-layout via per-wave LDS tile
#pragma unroll
        for (int r = 0; r < 8; r++) {
            pbuf[wv][r + 8 * hi][nc]      = (_Float16)p0[r];
            pbuf[wv][r + 8 * hi][16 + nc] = (_Float16)p1[r];
        }
        asm volatile("s_wait_dscnt 0" ::: "memory");
        v16h pa = frag_a(&pbuf[wv][0][0], 32);
        o0 = WMMA(pa, frag_bt(&vbuf[0][0], 32), o0);
        o1 = WMMA(pa, frag_bt(&vbuf[16][0], 32), o1);
        o2 = WMMA(pa, frag_bt(&vbuf[32][0], 32), o2);
        o3 = WMMA(pa, frag_bt(&vbuf[48][0], 32), o3);
        __syncthreads();  // qbuf/vbuf reused next chunk
    }

    _Float16* out = Oh + (size_t)bh * NPOS * DHEAD;
#pragma unroll
    for (int r = 0; r < 8; r++) {
        float inv = 1.0f / lrun[r];
        size_t o = (size_t)irow[r] * DHEAD + nc;
        out[o + 0]  = (_Float16)(o0[r] * inv);
        out[o + 16] = (_Float16)(o1[r] * inv);
        out[o + 32] = (_Float16)(o2[r] * inv);
        out[o + 48] = (_Float16)(o3[r] * inv);
    }
}

// ---- Per-head output projection + fused BatchNorm -> concat buffer ---------
__global__ __launch_bounds__(256) void head_proj(
    const _Float16* __restrict__ Oh, const _Float16* __restrict__ WoT,
    const float* __restrict__ bnscale, const float* __restrict__ bnbias,
    _Float16* __restrict__ cat) {
    const int wv = threadIdx.x >> 5;
    const int lane = threadIdx.x & 31;
    const int nc = lane & 15, hi = lane >> 4;
    const int bh = blockIdx.x >> 3;
    const int rowbase = (blockIdx.x & 7) * 128 + wv * 16;
    const int b = bh >> 3, h = bh & 7;

    const _Float16* A = Oh + (size_t)bh * NPOS * DHEAD;
    const _Float16* BT = WoT + (size_t)h * HOUT * DHEAD;

    v8f acc[4] = {v8f{}, v8f{}, v8f{}, v8f{}};
    for (int k = 0; k < DHEAD; k += 32) {
        v16h a = frag_a(A + (size_t)rowbase * DHEAD + k, DHEAD);
#pragma unroll
        for (int t = 0; t < 4; t++)
            acc[t] = WMMA(a, frag_bt(BT + (size_t)(t * 16) * DHEAD + k, DHEAD), acc[t]);
    }
#pragma unroll
    for (int t = 0; t < 4; t++) {
        int e = t * 16 + nc;
        float sc = bnscale[h * HOUT + e];
        float bi = bnbias[h * HOUT + e];
#pragma unroll
        for (int r = 0; r < 8; r++) {
            size_t row = (size_t)b * NPOS + rowbase + r + 8 * hi;
            cat[row * (NHEAD * HOUT) + h * HOUT + e] = (_Float16)(acc[t][r] * sc + bi);
        }
    }
}

// ---- Final 1x1 conv (8192x512 @ 512x512) + residual, f32 output ------------
__global__ __launch_bounds__(256) void final_gemm(
    const _Float16* __restrict__ cat, const _Float16* __restrict__ WoutT,
    const float* __restrict__ x, float* __restrict__ y) {
    const int wv = threadIdx.x >> 5;
    const int lane = threadIdx.x & 31;
    const int nc = lane & 15, hi = lane >> 4;
    const int rowblk = blockIdx.x & 63;
    const int colblk = blockIdx.x >> 6;
    const int rowbase = rowblk * 128 + wv * 16;
    const int colbase = colblk * 128;

    v8f acc[8] = {v8f{}, v8f{}, v8f{}, v8f{}, v8f{}, v8f{}, v8f{}, v8f{}};
    for (int k = 0; k < COUT; k += 32) {
        v16h a = frag_a(cat + (size_t)rowbase * COUT + k, COUT);
#pragma unroll
        for (int t = 0; t < 8; t++)
            acc[t] = WMMA(a, frag_bt(WoutT + (size_t)(colbase + t * 16) * COUT + k, COUT), acc[t]);
    }
#pragma unroll
    for (int t = 0; t < 8; t++)
#pragma unroll
        for (int r = 0; r < 8; r++) {
            size_t row = rowbase + r + 8 * hi;
            size_t c = colbase + t * 16 + nc;
            y[row * COUT + c] = acc[t][r] + x[row * COUT + c];
        }
}

// ---------------------------------------------------------------------------
extern "C" void kernel_launch(void* const* d_in, const int* in_sizes, int n_in,
                              void* d_out, int out_size, void* d_ws, size_t ws_size,
                              hipStream_t stream) {
    const float* x       = (const float*)d_in[0];
    const float* Wk      = (const float*)d_in[1];
    const float* Wq      = (const float*)d_in[2];
    const float* Wv      = (const float*)d_in[3];
    const float* Wo      = (const float*)d_in[4];
    const float* gamma   = (const float*)d_in[5];
    const float* beta    = (const float*)d_in[6];
    const float* movmean = (const float*)d_in[7];
    const float* movvar  = (const float*)d_in[8];
    const float* rel_w   = (const float*)d_in[9];
    const float* rel_h   = (const float*)d_in[10];
    const float* Wout    = (const float*)d_in[11];
    float* y = (float*)d_out;

    // Workspace carve-up (all 256B aligned)
    char* ws = (char*)d_ws;
    size_t off = 0;
    auto take = [&](size_t bytes) -> char* {
        char* p = ws + off;
        off += (bytes + 255) & ~(size_t)255;
        return p;
    };
    _Float16* xh    = (_Float16*)take((size_t)BATCH * NPOS * CIN * 2);
    _Float16* Qh    = (_Float16*)take((size_t)BATCH * NHEAD * NPOS * DHEAD * 2);
    _Float16* Kh    = (_Float16*)take((size_t)BATCH * NHEAD * NPOS * DHEAD * 2);
    _Float16* Vt    = (_Float16*)take((size_t)BATCH * NHEAD * DHEAD * NPOS * 2);
    _Float16* Oh    = (_Float16*)take((size_t)BATCH * NHEAD * NPOS * DHEAD * 2);
    _Float16* cat   = (_Float16*)take((size_t)BATCH * NPOS * NHEAD * HOUT * 2);
    _Float16* WqT   = (_Float16*)take((size_t)NHEAD * DHEAD * CIN * 2);
    _Float16* WkT   = (_Float16*)take((size_t)NHEAD * DHEAD * CIN * 2);
    _Float16* WvT   = (_Float16*)take((size_t)NHEAD * DHEAD * CIN * 2);
    _Float16* WoT   = (_Float16*)take((size_t)NHEAD * HOUT * DHEAD * 2);
    _Float16* WoutT = (_Float16*)take((size_t)COUT * (NHEAD * HOUT) * 2);
    float* bnscale  = (float*)take((size_t)NHEAD * HOUT * 4);
    float* bnbias   = (float*)take((size_t)NHEAD * HOUT * 4);
    float* wabs     = (float*)take((size_t)BATCH * NHEAD * NPOS * 32 * 4);
    float* habs     = (float*)take((size_t)BATCH * NHEAD * NPOS * 32 * 4);
    (void)ws_size; (void)in_sizes; (void)n_in; (void)out_size;

    const int T = 256;
    cvt_f32_f16<<<(BATCH * NPOS * CIN + T - 1) / T, T, 0, stream>>>(x, xh, BATCH * NPOS * CIN);
    transpose_cd<<<(NHEAD * CIN * DHEAD + T - 1) / T, T, 0, stream>>>(Wq, WqT, NHEAD, CIN, DHEAD);
    transpose_cd<<<(NHEAD * CIN * DHEAD + T - 1) / T, T, 0, stream>>>(Wk, WkT, NHEAD, CIN, DHEAD);
    transpose_cd<<<(NHEAD * CIN * DHEAD + T - 1) / T, T, 0, stream>>>(Wv, WvT, NHEAD, CIN, DHEAD);
    transpose_cd<<<(NHEAD * DHEAD * HOUT + T - 1) / T, T, 0, stream>>>(Wo, WoT, NHEAD, DHEAD, HOUT);
    transpose_cd<<<(COUT * COUT + T - 1) / T, T, 0, stream>>>(Wout, WoutT, 1, NHEAD * HOUT, COUT);
    bn_prep<<<2, T, 0, stream>>>(gamma, beta, movmean, movvar, bnscale, bnbias);
    qkv_proj<<<BATCH * NHEAD * 3 * 8, T, 0, stream>>>(xh, WqT, WkT, WvT, Qh, Kh, Vt);
    rel_logits<<<(BATCH * NHEAD * NPOS * 32 + T - 1) / T, T, 0, stream>>>(Qh, rel_w, wabs, 0);
    rel_logits<<<(BATCH * NHEAD * NPOS * 32 + T - 1) / T, T, 0, stream>>>(Qh, rel_h, habs, 1);
    attn_flash<<<BATCH * NHEAD * 8, T, 0, stream>>>(Qh, Kh, Vt, wabs, habs, Oh);
    head_proj<<<BATCH * NHEAD * 8, T, 0, stream>>>(Oh, WoT, bnscale, bnbias, cat);
    final_gemm<<<256, T, 0, stream>>>(cat, WoutT, x, y);
}
